// HymbaFlexAttention_61246233641059
// MI455X (gfx1250) — matrix operand order repaired and checked by reading, not compiled, not run on hardware
//
#include <hip/hip_runtime.h>
#include <hip/hip_bf16.h>

// ---------------- types ----------------
typedef __bf16 bf16_t;
typedef __attribute__((ext_vector_type(16))) __bf16 v16bf;
typedef __attribute__((ext_vector_type(8)))  float  v8f;
typedef __attribute__((ext_vector_type(4)))  unsigned int u32x4;
typedef __attribute__((ext_vector_type(2)))  unsigned int u32x2;
typedef __attribute__((ext_vector_type(4)))  float  f32x4;

union Frag { v16bf v; u32x4 u[2]; };

#define WMMA_BF16(A, B, C) \
  __builtin_amdgcn_wmma_f32_16x16x32_bf16(false, (A), false, (B), (short)0, (C), false, false)

// ---------------- problem constants ----------------
#define DMODEL 2048
#define NH_Q   16
#define NH_KV  4
#define DHEAD  128
#define WINSZ  256
#define MMEM   128
#define BB     4
#define TT     4096
#define TK     384          // MMEM + min(WIN, T-M)
#define KV_SHIFT 3712       // (T - w) - M = 3840 - 128

// ln(10000)/128
#define ROPE_LOG_OVER_DH 0.07195578468628f
#define SOFTMAX_SCALE    0.08838834764831845f   // 1/sqrt(128)

// ---------------- fp32 -> bf16 convert (4-wide) ----------------
__global__ __launch_bounds__(256) void k_cvt(const float* __restrict__ s,
                                             bf16_t* __restrict__ d, int n4) {
  int i = blockIdx.x * 256 + threadIdx.x;
  int stride = gridDim.x * 256;
  for (; i < n4; i += stride) {
    f32x4 f = ((const f32x4*)s)[i];
    union { bf16_t h[4]; u32x2 u; } pk;
    pk.h[0] = (bf16_t)f.x; pk.h[1] = (bf16_t)f.y;
    pk.h[2] = (bf16_t)f.z; pk.h[3] = (bf16_t)f.w;
    ((u32x2*)d)[i] = pk.u;
  }
}

// ---------------- projection GEMM + fused RoPE ----------------
// out[g, n] = sum_k xb[row(g), k] * w[n, k]   (bf16 WMMA, fp32 acc)
// Each wave: 32 rows x 64 cols, double-buffered K loop of 32.
// mode 0: g = 0..B*T-1, row = g, pos = g % T        (Q)
// mode 1: g = 0..B*384-1, j = g%384, pos = j (+3712 if j>=128), row = b*T+pos (K/V)
// Output layout: ((b*NH + head)*R + rloc)*128 + dh  (bf16)
__global__ __launch_bounds__(256) void k_proj(const bf16_t* __restrict__ xb,
                                              const bf16_t* __restrict__ w,
                                              bf16_t* __restrict__ outp,
                                              int NH, int R, int mode, int rope) {
  const int lane = threadIdx.x & 31;
  const int wv   = threadIdx.x >> 5;
  const int n16  = lane & 15;
  const int half = lane >> 4;
  const int mrow = half * 8;     // C/D row base for this lane group
  const int akb  = half * 16;    // A-fragment K base
  const int bkb  = half * 16;    // B-fragment K base

  const int c0    = blockIdx.x * 64;
  const int rbase = blockIdx.y * 256 + wv * 32;   // 32 rows per wave

  // A rows for this lane's two fragments (m = lane%16, row groups +0 and +16)
  const bf16_t* aptr[2];
#pragma unroll
  for (int i = 0; i < 2; ++i) {
    int g_a = rbase + i * 16 + n16;
    long xrow;
    if (mode == 0) {
      xrow = g_a;
    } else {
      int b = g_a / TK, j = g_a % TK;
      int pos = j + (j >= MMEM ? KV_SHIFT : 0);
      xrow = (long)b * TT + pos;
    }
    aptr[i] = xb + xrow * DMODEL;
  }

  const bf16_t* bptr[4];
#pragma unroll
  for (int j = 0; j < 4; ++j) bptr[j] = w + (long)(c0 + j * 16 + n16) * DMODEL;

  v8f acc[2][4];
#pragma unroll
  for (int i = 0; i < 2; ++i)
#pragma unroll
    for (int j = 0; j < 4; ++j) acc[i][j] = (v8f){};

  Frag a0, a1, b0, b1, b2, b3;        // stage X
  Frag c0f, c1f, d0, d1, d2, d3;      // stage Y

#define LD_SET(A0, A1, B0, B1, B2, B3, K)                                \
  do {                                                                   \
    (A0).u[0] = *(const u32x4*)(aptr[0] + (K) + akb);                    \
    (A0).u[1] = *(const u32x4*)(aptr[0] + (K) + akb + 8);                \
    (A1).u[0] = *(const u32x4*)(aptr[1] + (K) + akb);                    \
    (A1).u[1] = *(const u32x4*)(aptr[1] + (K) + akb + 8);                \
    (B0).u[0] = *(const u32x4*)(bptr[0] + (K) + bkb);                    \
    (B0).u[1] = *(const u32x4*)(bptr[0] + (K) + bkb + 8);                \
    (B1).u[0] = *(const u32x4*)(bptr[1] + (K) + bkb);                    \
    (B1).u[1] = *(const u32x4*)(bptr[1] + (K) + bkb + 8);                \
    (B2).u[0] = *(const u32x4*)(bptr[2] + (K) + bkb);                    \
    (B2).u[1] = *(const u32x4*)(bptr[2] + (K) + bkb + 8);                \
    (B3).u[0] = *(const u32x4*)(bptr[3] + (K) + bkb);                    \
    (B3).u[1] = *(const u32x4*)(bptr[3] + (K) + bkb + 8);                \
  } while (0)

#define MMA_SET(A0, A1, B0, B1, B2, B3)                 \
  do {                                                  \
    acc[0][0] = WMMA_BF16((A0).v, (B0).v, acc[0][0]);   \
    acc[0][1] = WMMA_BF16((A0).v, (B1).v, acc[0][1]);   \
    acc[0][2] = WMMA_BF16((A0).v, (B2).v, acc[0][2]);   \
    acc[0][3] = WMMA_BF16((A0).v, (B3).v, acc[0][3]);   \
    acc[1][0] = WMMA_BF16((A1).v, (B0).v, acc[1][0]);   \
    acc[1][1] = WMMA_BF16((A1).v, (B1).v, acc[1][1]);   \
    acc[1][2] = WMMA_BF16((A1).v, (B2).v, acc[1][2]);   \
    acc[1][3] = WMMA_BF16((A1).v, (B3).v, acc[1][3]);   \
  } while (0)

  LD_SET(a0, a1, b0, b1, b2, b3, 0);
  for (int k0 = 0; k0 < DMODEL; k0 += 64) {
    __builtin_prefetch((const void*)(aptr[0] + k0 + 512), 0, 3);
    // stage Y loads (k0+32) overlap stage X WMMAs
    LD_SET(c0f, c1f, d0, d1, d2, d3, k0 + 32);
    MMA_SET(a0, a1, b0, b1, b2, b3);
    // stage X loads for next iteration (clamped in-bounds; unused on last)
    int kn = (k0 + 64 < DMODEL) ? k0 + 64 : 0;
    LD_SET(a0, a1, b0, b1, b2, b3, kn);
    MMA_SET(c0f, c1f, d0, d1, d2, d3);
  }
#undef LD_SET
#undef MMA_SET

  // epilogue: fused RoPE + bf16 store
#pragma unroll
  for (int i = 0; i < 2; ++i) {
#pragma unroll
    for (int j = 0; j < 4; ++j) {
      int ncol = c0 + j * 16 + n16;
      int head = ncol / DHEAD;
      int dh   = ncol % DHEAD;
      float sgn  = (dh & 1) ? 1.0f : -1.0f;
      float invf = __expf(-(float)(dh & ~1) * ROPE_LOG_OVER_DH);
#pragma unroll
      for (int r = 0; r < 8; ++r) {
        float self  = acc[i][j][r];
        float other = __shfl_xor(self, 1, 32);   // even<->odd feature partner
        int g = rbase + i * 16 + mrow + r;
        int bidx, rloc, pos;
        if (mode == 0) { bidx = g >> 12; rloc = g & (TT - 1); pos = rloc; }
        else { bidx = g / TK; rloc = g % TK; pos = rloc + (rloc >= MMEM ? KV_SHIFT : 0); }
        float val = self;
        if (rope) {
          float ang = (float)pos * invf;
          float sa, ca;
          __sincosf(ang, &sa, &ca);
          val = self * ca + sgn * other * sa;
        }
        outp[(((long)bidx * NH + head) * R + rloc) * DHEAD + dh] = (bf16_t)val;
      }
    }
  }
}

// ---------------- attention (flash-style, bf16 WMMA) ----------------
// grid: (T/128, H, B); block 256 (8 waves), each wave owns 16 query rows.
// LDS: Ks[384][128] row-major, Vt[128][384] transposed, Ps[8][16*32] staging.
__global__ __launch_bounds__(256) void k_attn(const bf16_t* __restrict__ qb,
                                              const bf16_t* __restrict__ kb,
                                              const bf16_t* __restrict__ vb,
                                              bf16_t* __restrict__ ob) {
  extern __shared__ __align__(16) char smem[];
  bf16_t* Ks = (bf16_t*)smem;             // 384*128
  bf16_t* Vt = Ks + TK * DHEAD;           // 128*384
  bf16_t* Ps = Vt + DHEAD * TK;           // 8 * 512

  const int b = blockIdx.z, h = blockIdx.y;
  const int wv   = threadIdx.x >> 5;
  const int lane = threadIdx.x & 31;
  const int n16  = lane & 15;
  const int half = lane >> 4;
  const int mrow = half * 8;
  const int akb  = half * 16;
  const int bkb  = half * 16;
  const int q0   = blockIdx.x * 128 + wv * 16;

  const bf16_t* kbp = kb + ((long)(b * NH_KV + (h >> 2)) * TK) * DHEAD;
  const bf16_t* vbp = vb + ((long)(b * NH_KV + (h >> 2)) * TK) * DHEAD;

  // cooperative K copy (row-major) : 6144 x 16B chunks
  for (int idx = threadIdx.x; idx < TK * DHEAD / 8; idx += 256)
    ((u32x4*)Ks)[idx] = ((const u32x4*)kbp)[idx];
  // cooperative V transpose into Vt[dh][key]
  for (int idx = threadIdx.x; idx < TK * DHEAD / 8; idx += 256) {
    int key = idx >> 4;          // 16 chunks per 128-wide row
    int dh0 = (idx & 15) * 8;
    u32x4 dv = ((const u32x4*)vbp)[idx];
    const bf16_t* e = (const bf16_t*)&dv;
#pragma unroll
    for (int i = 0; i < 8; ++i) Vt[(dh0 + i) * TK + key] = e[i];
  }
  __syncthreads();

  // Q fragments for this wave's 16 rows (DH=128 -> 4 K-steps of 32)
  const bf16_t* qptr = qb + (((long)(b * NH_Q + h) * TT) + q0 + n16) * DHEAD;
  Frag qf[4];
#pragma unroll
  for (int ks = 0; ks < 4; ++ks) {
    qf[ks].u[0] = *(const u32x4*)(qptr + ks * 32 + akb);
    qf[ks].u[1] = *(const u32x4*)(qptr + ks * 32 + akb + 8);
  }

  float mst[8], lst[8];
  v8f Oacc[8];
#pragma unroll
  for (int r = 0; r < 8; ++r) { mst[r] = -3.0e38f; lst[r] = 0.0f; }
#pragma unroll
  for (int nt = 0; nt < 8; ++nt) Oacc[nt] = (v8f){};

  bf16_t* Pw = Ps + wv * 512;   // per-wave 16x32 staging

  for (int dt = 0; dt < TK / 32; ++dt) {     // 12 double-tiles of 32 keys
    v8f sv[2];
#pragma unroll
    for (int hlf = 0; hlf < 2; ++hlf) {
      int kt = dt * 2 + hlf;
      // preload all 4 K B-fragments (ds-load clause), then 4 WMMAs
      Frag kf[4];
#pragma unroll
      for (int ks = 0; ks < 4; ++ks) {
        const bf16_t* kp = Ks + (kt * 16 + n16) * DHEAD + ks * 32 + bkb;
        kf[ks].u[0] = *(const u32x4*)kp;
        kf[ks].u[1] = *(const u32x4*)(kp + 8);
      }
      v8f s = {};
#pragma unroll
      for (int ks = 0; ks < 4; ++ks) s = WMMA_BF16(qf[ks].v, kf[ks].v, s);

      int j = kt * 16 + n16;     // slot index (reference masks on slot index)
#pragma unroll
      for (int r = 0; r < 8; ++r) {
        int qi = q0 + mrow + r;
        bool ok = (j < MMEM) || ((qi >= j) && (qi - j < WINSZ));
        s[r] = ok ? s[r] * SOFTMAX_SCALE : -3.0e38f;
      }
      sv[hlf] = s;
    }

    float p[2][8];
#pragma unroll
    for (int r = 0; r < 8; ++r) {
      float mx = fmaxf(sv[0][r], sv[1][r]);
#pragma unroll
      for (int d = 8; d >= 1; d >>= 1) mx = fmaxf(mx, __shfl_xor(mx, d, 32));
      float newm  = fmaxf(mst[r], mx);
      float alpha = __expf(mst[r] - newm);
      float p0 = __expf(sv[0][r] - newm);
      float p1 = __expf(sv[1][r] - newm);
      float rs = p0 + p1;
#pragma unroll
      for (int d = 8; d >= 1; d >>= 1) rs += __shfl_xor(rs, d, 32);
      lst[r] = lst[r] * alpha + rs;
      mst[r] = newm;
      p[0][r] = p0; p[1][r] = p1;
#pragma unroll
      for (int nt = 0; nt < 8; ++nt) Oacc[nt][r] *= alpha;
    }

    // transpose P through per-wave LDS staging -> A-fragment layout
#pragma unroll
    for (int hlf = 0; hlf < 2; ++hlf)
#pragma unroll
      for (int r = 0; r < 8; ++r)
        Pw[(mrow + r) * 32 + hlf * 16 + n16] = (bf16_t)p[hlf][r];
    asm volatile("s_wait_dscnt 0" ::: "memory");
    Frag pa;
    pa.u[0] = *(const u32x4*)(Pw + n16 * 32 + akb);
    pa.u[1] = *(const u32x4*)(Pw + n16 * 32 + akb + 8);

    // O += P (16x32) @ V (32x128): two groups of 4 preloaded V fragments
#pragma unroll
    for (int g = 0; g < 2; ++g) {
      Frag vf[4];
#pragma unroll
      for (int t = 0; t < 4; ++t) {
        const bf16_t* vp = Vt + ((g * 4 + t) * 16 + n16) * TK + dt * 32 + bkb;
        vf[t].u[0] = *(const u32x4*)vp;
        vf[t].u[1] = *(const u32x4*)(vp + 8);
      }
#pragma unroll
      for (int t = 0; t < 4; ++t)
        Oacc[g * 4 + t] = WMMA_BF16(pa.v, vf[t].v, Oacc[g * 4 + t]);
    }
  }

  // normalize and store to (B, T, H*DH) bf16
#pragma unroll
  for (int nt = 0; nt < 8; ++nt)
#pragma unroll
    for (int r = 0; r < 8; ++r) {
      float o = Oacc[nt][r] / lst[r];
      ob[((long)(b * TT) + q0 + mrow + r) * (NH_Q * DHEAD)
         + h * DHEAD + nt * 16 + n16] = (bf16_t)o;
    }
}

// ---------------- output projection: out(fp32) = ob @ wo^T ----------------
// Each wave: 32 rows x 64 cols, double-buffered K loop.
__global__ __launch_bounds__(256) void k_oproj(const bf16_t* __restrict__ ab,
                                               const bf16_t* __restrict__ w,
                                               float* __restrict__ outp) {
  const int lane = threadIdx.x & 31;
  const int wv   = threadIdx.x >> 5;
  const int n16  = lane & 15;
  const int half = lane >> 4;
  const int mrow = half * 8;
  const int akb  = half * 16;
  const int bkb  = half * 16;

  const int c0    = blockIdx.x * 64;
  const int rbase = blockIdx.y * 256 + wv * 32;

  const bf16_t* aptr[2];
  aptr[0] = ab + (long)(rbase + n16) * DMODEL;
  aptr[1] = ab + (long)(rbase + 16 + n16) * DMODEL;
  const bf16_t* bptr[4];
#pragma unroll
  for (int j = 0; j < 4; ++j) bptr[j] = w + (long)(c0 + j * 16 + n16) * DMODEL;

  v8f acc[2][4];
#pragma unroll
  for (int i = 0; i < 2; ++i)
#pragma unroll
    for (int j = 0; j < 4; ++j) acc[i][j] = (v8f){};

  Frag a0, a1, b0, b1, b2, b3;
  Frag c0f, c1f, d0, d1, d2, d3;

#define LD_SET(A0, A1, B0, B1, B2, B3, K)                                \
  do {                                                                   \
    (A0).u[0] = *(const u32x4*)(aptr[0] + (K) + akb);                    \
    (A0).u[1] = *(const u32x4*)(aptr[0] + (K) + akb + 8);                \
    (A1).u[0] = *(const u32x4*)(aptr[1] + (K) + akb);                    \
    (A1).u[1] = *(const u32x4*)(aptr[1] + (K) + akb + 8);                \
    (B0).u[0] = *(const u32x4*)(bptr[0] + (K) + bkb);                    \
    (B0).u[1] = *(const u32x4*)(bptr[0] + (K) + bkb + 8);                \
    (B1).u[0] = *(const u32x4*)(bptr[1] + (K) + bkb);                    \
    (B1).u[1] = *(const u32x4*)(bptr[1] + (K) + bkb + 8);                \
    (B2).u[0] = *(const u32x4*)(bptr[2] + (K) + bkb);                    \
    (B2).u[1] = *(const u32x4*)(bptr[2] + (K) + bkb + 8);                \
    (B3).u[0] = *(const u32x4*)(bptr[3] + (K) + bkb);                    \
    (B3).u[1] = *(const u32x4*)(bptr[3] + (K) + bkb + 8);                \
  } while (0)

#define MMA_SET(A0, A1, B0, B1, B2, B3)                 \
  do {                                                  \
    acc[0][0] = WMMA_BF16((A0).v, (B0).v, acc[0][0]);   \
    acc[0][1] = WMMA_BF16((A0).v, (B1).v, acc[0][1]);   \
    acc[0][2] = WMMA_BF16((A0).v, (B2).v, acc[0][2]);   \
    acc[0][3] = WMMA_BF16((A0).v, (B3).v, acc[0][3]);   \
    acc[1][0] = WMMA_BF16((A1).v, (B0).v, acc[1][0]);   \
    acc[1][1] = WMMA_BF16((A1).v, (B1).v, acc[1][1]);   \
    acc[1][2] = WMMA_BF16((A1).v, (B2).v, acc[1][2]);   \
    acc[1][3] = WMMA_BF16((A1).v, (B3).v, acc[1][3]);   \
  } while (0)

  LD_SET(a0, a1, b0, b1, b2, b3, 0);
  for (int k0 = 0; k0 < DMODEL; k0 += 64) {
    __builtin_prefetch((const void*)(aptr[0] + k0 + 512), 0, 3);
    LD_SET(c0f, c1f, d0, d1, d2, d3, k0 + 32);
    MMA_SET(a0, a1, b0, b1, b2, b3);
    int kn = (k0 + 64 < DMODEL) ? k0 + 64 : 0;
    LD_SET(a0, a1, b0, b1, b2, b3, kn);
    MMA_SET(c0f, c1f, d0, d1, d2, d3);
  }
#undef LD_SET
#undef MMA_SET

#pragma unroll
  for (int i = 0; i < 2; ++i)
#pragma unroll
    for (int j = 0; j < 4; ++j)
#pragma unroll
      for (int r = 0; r < 8; ++r)
        outp[(long)(rbase + i * 16 + mrow + r) * DMODEL + c0 + j * 16 + n16] =
            acc[i][j][r];
}

// ---------------- host launch ----------------
extern "C" void kernel_launch(void* const* d_in, const int* in_sizes, int n_in,
                              void* d_out, int out_size, void* d_ws, size_t ws_size,
                              hipStream_t stream) {
  (void)in_sizes; (void)n_in; (void)out_size; (void)ws_size;
  const float* x  = (const float*)d_in[0];
  const float* wq = (const float*)d_in[1];
  const float* wk = (const float*)d_in[2];
  const float* wv = (const float*)d_in[3];
  const float* wo = (const float*)d_in[4];
  float* out = (float*)d_out;

  char* ws = (char*)d_ws;
  size_t off = 0;
  auto carve = [&](size_t bytes) -> void* {
    void* p = ws + off;
    off += (bytes + 255) & ~(size_t)255;
    return p;
  };
  const size_t NX  = (size_t)BB * TT * DMODEL;            // 33.5M
  const size_t NWQ = (size_t)NH_Q * DHEAD * DMODEL;       // 4.19M
  const size_t NWK = (size_t)NH_KV * DHEAD * DMODEL;      // 1.05M
  const size_t NQ  = (size_t)BB * NH_Q * TT * DHEAD;      // 33.5M
  const size_t NKV = (size_t)BB * NH_KV * TK * DHEAD;     // 786K

  bf16_t* xb  = (bf16_t*)carve(sizeof(bf16_t) * NX);
  bf16_t* wqb = (bf16_t*)carve(sizeof(bf16_t) * NWQ);
  bf16_t* wkb = (bf16_t*)carve(sizeof(bf16_t) * NWK);
  bf16_t* wvb = (bf16_t*)carve(sizeof(bf16_t) * NWK);
  bf16_t* wob = (bf16_t*)carve(sizeof(bf16_t) * NWQ);
  bf16_t* qb  = (bf16_t*)carve(sizeof(bf16_t) * NQ);
  bf16_t* kbf = (bf16_t*)carve(sizeof(bf16_t) * NKV);
  bf16_t* vbf = (bf16_t*)carve(sizeof(bf16_t) * NKV);
  bf16_t* obf = (bf16_t*)carve(sizeof(bf16_t) * NX);

  auto cvt = [&](const float* s, bf16_t* d, size_t n) {
    int n4 = (int)(n / 4);
    int grid = (n4 + 255) / 256;
    k_cvt<<<grid, 256, 0, stream>>>(s, d, n4);
  };
  cvt(x, xb, NX);
  cvt(wq, wqb, NWQ);
  cvt(wk, wkb, NWK);
  cvt(wv, wvb, NWK);
  cvt(wo, wob, NWQ);

  // Q projection + RoPE: 16384 rows x 2048 cols (wave tile 32x64)
  k_proj<<<dim3(DMODEL / 64, (BB * TT) / 256), 256, 0, stream>>>(
      xb, wqb, qb, NH_Q, TT, /*mode=*/0, /*rope=*/1);
  // K projection + RoPE at original positions: B*384 rows x 512 cols
  k_proj<<<dim3((NH_KV * DHEAD) / 64, (BB * TK) / 256), 256, 0, stream>>>(
      xb, wkb, kbf, NH_KV, TK, /*mode=*/1, /*rope=*/1);
  // V projection (no RoPE)
  k_proj<<<dim3((NH_KV * DHEAD) / 64, (BB * TK) / 256), 256, 0, stream>>>(
      xb, wvb, vbf, NH_KV, TK, /*mode=*/1, /*rope=*/0);

  // attention
  size_t smem_attn = (size_t)(TK * DHEAD + DHEAD * TK + 8 * 512) * sizeof(bf16_t); // 204800 B
  hipFuncSetAttribute((const void*)k_attn,
                      hipFuncAttributeMaxDynamicSharedMemorySize, (int)smem_attn);
  k_attn<<<dim3(TT / 128, NH_Q, BB), 256, smem_attn, stream>>>(qb, kbf, vbf, obf);

  // output projection (fp32 result)
  k_oproj<<<dim3(DMODEL / 64, (BB * TT) / 256), 256, 0, stream>>>(obf, wob, out);
}
